// NemotronHMTP_11364483465232
// MI455X (gfx1250) — matrix operand order repaired and checked by reading, not compiled, not run on hardware
//
#include <hip/hip_runtime.h>
#include <hip/hip_bf16.h>

// ---------------- problem constants ----------------
constexpr int T_    = 2048;
constexpr int D_    = 2048;
constexpr int DL_   = 1024;
constexpr int E_    = 64;
constexpr int I_    = 768;
constexpr int SI_   = 4096;
constexpr int TOPK_ = 6;
constexpr int CAP_  = 384;
constexpr float ROUTED_SCALE_ = 2.5f;

typedef __bf16 bf16_t;
typedef __attribute__((ext_vector_type(16))) __bf16 v16bf;
typedef __attribute__((ext_vector_type(8)))  float  v8f;

#define LDSS 40   // padded LDS row stride (elements); 80B keeps 16B alignment

enum { EPI_F32 = 0, EPI_BF16 = 1, EPI_RELU2_BF16 = 2, EPI_ADD_F32 = 3 };

// ============================================================
// C[M,N] = A[M,K](bf16) x B[N,K]^T(f32, converted inline)
// Block tile 128(M) x 64(N), 128 threads (4 waves).
// Each wave computes 32(M) x 64(N): 2 a-frags x 4 b-frags
//   -> 8 v_wmma_f32_16x16x32_bf16 per K-step.
// Software pipeline: A tile k+1 via global_load_async_to_lds_b128
// (CDNA5 async DMA, ASYNCcnt), B tile k+1 via VGPR prefetch with
// inline f32->bf16 conversion; double-buffered LDS, one barrier
// per K-step. M mult of 128, N mult of 64, K mult of 32.
// ============================================================
template<int EPI>
__global__ __launch_bounds__(128)
void gemm_xwt(const bf16_t* __restrict__ A, const float* __restrict__ Bw,
              void* __restrict__ C, int M, int N, int K,
              long long batchA, long long batchB, long long batchC,
              const int* __restrict__ cnt)
{
    const int bz = blockIdx.z;
    if (cnt && (int)(blockIdx.y * 128) >= cnt[bz]) return;   // skip empty expert tiles
    A  += (size_t)bz * (size_t)batchA;
    Bw += (size_t)bz * (size_t)batchB;

    __shared__ bf16_t As[2][128 * LDSS];
    __shared__ bf16_t Bs[2][64 * LDSS];

    const int tid   = threadIdx.x;
    const int lane  = tid & 31;
    const int wv    = tid >> 5;
    const int wm    = wv * 32;            // wave's M offset within block tile
    const int mBase = blockIdx.y * 128;
    const int nBase = blockIdx.x * 64;

    // ---- async A staging: 512 16B chunks per 128x32 tile, 4 per thread ----
    auto asyncA = [&](int kb, int pbuf) {
        #pragma unroll
        for (int it = 0; it < 4; ++it) {
            int chunk = tid + it * 128;
            int row = chunk >> 2;
            int c8  = (chunk & 3) * 8;
            const bf16_t* g = A + (size_t)(mBase + row) * K + kb + c8;
            unsigned lds = (unsigned)(uintptr_t)(&As[pbuf][row * LDSS + c8]);
            asm volatile("global_load_async_to_lds_b128 %0, %1, off"
                         :: "v"(lds), "v"(g) : "memory");
        }
    };
    // ---- B staging through VGPRs (f32 -> bf16): 64x32 tile ----
    float4 br[2][2];
    auto loadB = [&](int kb) {
        #pragma unroll
        for (int it = 0; it < 2; ++it) {
            int chunk = tid + it * 128;
            int row = chunk >> 2;
            int c8  = (chunk & 3) * 8;
            const float4* s = (const float4*)(Bw + (size_t)(nBase + row) * K + kb + c8);
            br[it][0] = s[0];
            br[it][1] = s[1];
        }
    };
    auto storeB = [&](int pbuf) {
        #pragma unroll
        for (int it = 0; it < 2; ++it) {
            int chunk = tid + it * 128;
            int row = chunk >> 2;
            int c8  = (chunk & 3) * 8;
            float4 f0 = br[it][0], f1 = br[it][1];
            union { bf16_t h[8]; uint4 u; } cv;
            cv.h[0] = (bf16_t)f0.x; cv.h[1] = (bf16_t)f0.y;
            cv.h[2] = (bf16_t)f0.z; cv.h[3] = (bf16_t)f0.w;
            cv.h[4] = (bf16_t)f1.x; cv.h[5] = (bf16_t)f1.y;
            cv.h[6] = (bf16_t)f1.z; cv.h[7] = (bf16_t)f1.w;
            *(uint4*)(&Bs[pbuf][row * LDSS + c8]) = cv.u;
        }
    };

    v8f acc[2][4];
    #pragma unroll
    for (int i = 0; i < 2; ++i)
        #pragma unroll
        for (int j = 0; j < 4; ++j) acc[i][j] = {};

    // ---- prologue: stage tile 0 into buffer 0 ----
    asyncA(0, 0);
    loadB(0);
    storeB(0);
    asm volatile("s_wait_asynccnt 0x0" ::: "memory");

    int p = 0;
    for (int kb = 0; kb < K; kb += 32) {
        __syncthreads();                 // publish LDS[p] (waits DScnt)
        const bool next = (kb + 32 < K);
        if (next) {                      // start tile k+1 while computing tile k
            asyncA(kb + 32, p ^ 1);
            loadB(kb + 32);
        }

        // ---- fragment loads, contiguous k order (k = 16*(lane/16) + s) for A and B ----
        const int mrow = lane & 15;
        const int khiA = (lane >> 4) * 16;   // A: lanes<16 K0..15, lanes>=16 K16..31
        const int khiB = (lane >> 4) * 16;   // B: lanes<16 K0..15, lanes>=16 K16..31
        union V16 { uint4 u[2]; v16bf v; };
        V16 a[2], b[4];
        #pragma unroll
        for (int i = 0; i < 2; ++i) {
            const bf16_t* r = &As[p][(wm + i * 16 + mrow) * LDSS];
            a[i].u[0] = *(const uint4*)(r + khiA);
            a[i].u[1] = *(const uint4*)(r + khiA + 8);
        }
        #pragma unroll
        for (int j = 0; j < 4; ++j) {
            const bf16_t* r = &Bs[p][(j * 16 + mrow) * LDSS];
            b[j].u[0] = *(const uint4*)(r + khiB);
            b[j].u[1] = *(const uint4*)(r + khiB + 8);
        }
        #pragma unroll
        for (int i = 0; i < 2; ++i)
            #pragma unroll
            for (int j = 0; j < 4; ++j)
                acc[i][j] = __builtin_amdgcn_wmma_f32_16x16x32_bf16(
                    false, a[i].v, false, b[j].v, (short)0, acc[i][j], false, false);

        if (next) {
            storeB(p ^ 1);
            asm volatile("s_wait_asynccnt 0x0" ::: "memory");  // A tile k+1 landed
        }
        p ^= 1;
    }

    // ---- epilogue (C/D layout: VGPR r -> M = r + 8*(lane>=16), N = lane&15) ----
    float*  Cpf = (float*)C  + (size_t)bz * (size_t)batchC;
    bf16_t* Cph = (bf16_t*)C + (size_t)bz * (size_t)batchC;
    const int rHi  = (lane >> 4) * 8;
    const int nCol = lane & 15;
    #pragma unroll
    for (int i = 0; i < 2; ++i)
    #pragma unroll
    for (int j = 0; j < 4; ++j)
    #pragma unroll
    for (int r = 0; r < 8; ++r) {
        int grow = mBase + wm + i * 16 + rHi + r;
        int gcol = nBase + j * 16 + nCol;
        size_t idx = (size_t)grow * N + gcol;
        float v = acc[i][j][r];
        if constexpr (EPI == EPI_F32)            { Cpf[idx] = v; }
        else if constexpr (EPI == EPI_ADD_F32)   { Cpf[idx] += v; }
        else if constexpr (EPI == EPI_BF16)      { Cph[idx] = (bf16_t)v; }
        else /* EPI_RELU2_BF16 */                { float rv = v > 0.f ? v : 0.f;
                                                   Cph[idx] = (bf16_t)(rv * rv); }
    }
}

// ---------------- f32 -> bf16 conversion (x) ----------------
__global__ __launch_bounds__(256)
void cvt_f32_bf16(const float* __restrict__ x, bf16_t* __restrict__ y, int n4)
{
    int i = blockIdx.x * 256 + threadIdx.x;
    if (i >= n4) return;
    float4 f = ((const float4*)x)[i];
    union { bf16_t h[4]; uint2 u; } cv;
    cv.h[0] = (bf16_t)f.x; cv.h[1] = (bf16_t)f.y;
    cv.h[2] = (bf16_t)f.z; cv.h[3] = (bf16_t)f.w;
    ((uint2*)y)[i] = cv.u;
}

// ---------------- routing (sigmoid + bias, group top-2 sums,
//                  top-4 groups, top-6 experts, normalized weights) --------
__global__ __launch_bounds__(256)
void routing_kernel(const float* __restrict__ logits, const float* __restrict__ bias,
                    int* __restrict__ tidx, float* __restrict__ tw, int* __restrict__ cnt)
{
    int t = blockIdx.x * 256 + threadIdx.x;
    if (blockIdx.x == 0 && threadIdx.x < E_) cnt[threadIdx.x] = 0;  // zero counters
    if (t >= T_) return;

    float sb[E_];
    #pragma unroll 8
    for (int e = 0; e < E_; ++e) {
        float l = logits[(size_t)t * E_ + e];
        float s = 1.f / (1.f + __expf(-l));
        sb[e] = s + bias[e];
    }
    // group scores = sum of top-2 within each group of 8
    float gs[8];
    #pragma unroll
    for (int g = 0; g < 8; ++g) {
        float m1 = -1e30f, m2 = -1e30f;
        #pragma unroll
        for (int j = 0; j < 8; ++j) {
            float v = sb[g * 8 + j];
            if (v > m1) { m2 = m1; m1 = v; } else if (v > m2) { m2 = v; }
        }
        gs[g] = m1 + m2;
    }
    // top-4 groups (strict > : ties pick lowest index, matching lax.top_k)
    unsigned gsel = 0;
    for (int it = 0; it < 4; ++it) {
        float best = -1e30f; int bg = 0;
        #pragma unroll
        for (int g = 0; g < 8; ++g)
            if (!((gsel >> g) & 1u) && gs[g] > best) { best = gs[g]; bg = g; }
        gsel |= 1u << bg;
    }
    // top-6 experts over masked scores (non-group value = 0.0, as reference)
    unsigned long long picked = 0;
    int   ids[TOPK_];
    float w[TOPK_];
    float wsum = 0.f;
    for (int it = 0; it < TOPK_; ++it) {
        float best = -1e30f; int bi = 0;
        for (int e = 0; e < E_; ++e) {
            if ((picked >> e) & 1ull) continue;
            float v = ((gsel >> (e >> 3)) & 1u) ? sb[e] : 0.0f;
            if (v > best) { best = v; bi = e; }
        }
        picked |= 1ull << bi;
        ids[it] = bi;
        float l = logits[(size_t)t * E_ + bi];
        float s = 1.f / (1.f + __expf(-l));   // weights from un-biased scores
        w[it] = s; wsum += s;
    }
    float inv = ROUTED_SCALE_ / (wsum + 1e-20f);
    #pragma unroll
    for (int k = 0; k < TOPK_; ++k) {
        tidx[(size_t)t * TOPK_ + k] = ids[k];
        tw[(size_t)t * TOPK_ + k]   = w[k] * inv;
    }
}

// ---------------- dispatch: slot assignment + row scatter ----------------
__global__ __launch_bounds__(64)
void dispatch_kernel(const int* __restrict__ tidx, const bf16_t* __restrict__ xl,
                     bf16_t* __restrict__ buf, int* __restrict__ cnt,
                     int* __restrict__ slot_of)
{
    int a = blockIdx.x;                   // (t,k) flat index, 0..T*K-1
    __shared__ int s_slot;
    if (threadIdx.x == 0) {
        int e = tidx[a];
        int pos = atomicAdd(&cnt[e], 1);
        int slot = (pos < CAP_) ? (e * CAP_ + pos) : -1;
        slot_of[a] = slot;
        s_slot = slot;
    }
    __syncthreads();
    int slot = s_slot;
    if (slot < 0) return;
    int t = a / TOPK_;
    const uint4* src = (const uint4*)(xl + (size_t)t * DL_);
    uint4*       dst = (uint4*)(buf + (size_t)slot * DL_);
    #pragma unroll
    for (int i = 0; i < 2; ++i)
        dst[threadIdx.x + i * 64] = src[threadIdx.x + i * 64];  // 2KB row
}

// ---------------- combine: weighted gather of 6 expert rows ----------------
__global__ __launch_bounds__(128)
void combine_kernel(const float* __restrict__ OUT, const int* __restrict__ slot_of,
                    const float* __restrict__ tw, bf16_t* __restrict__ routedb)
{
    int t = blockIdx.x;
    int c = threadIdx.x;                  // cols c, c+128, ..., c+896
    float acc[8] = {0.f, 0.f, 0.f, 0.f, 0.f, 0.f, 0.f, 0.f};
    for (int k = 0; k < TOPK_; ++k) {     // fixed order -> deterministic
        int s = slot_of[(size_t)t * TOPK_ + k];
        if (s < 0) continue;
        float wk = tw[(size_t)t * TOPK_ + k];
        const float* row = OUT + (size_t)s * DL_;
        #pragma unroll
        for (int j = 0; j < 8; ++j) acc[j] += wk * row[c + j * 128];
    }
    #pragma unroll
    for (int j = 0; j < 8; ++j)
        routedb[(size_t)t * DL_ + c + j * 128] = (bf16_t)acc[j];
}

// ============================================================
extern "C" void kernel_launch(void* const* d_in, const int* in_sizes, int n_in,
                              void* d_out, int out_size, void* d_ws, size_t ws_size,
                              hipStream_t stream)
{
    const float* x      = (const float*)d_in[0];
    const float* gate_w = (const float*)d_in[1];
    const float* gate_b = (const float*)d_in[2];
    const float* fc1_w  = (const float*)d_in[3];
    const float* fc2_w  = (const float*)d_in[4];
    const float* w1     = (const float*)d_in[5];
    const float* w2     = (const float*)d_in[6];
    const float* sw1    = (const float*)d_in[7];
    const float* sw2    = (const float*)d_in[8];
    float* out = (float*)d_out;

    // -------- workspace layout (256B aligned) --------
    char* ws = (char*)d_ws;
    size_t off = 0;
    auto alloc = [&](size_t bytes) -> char* {
        char* p = ws + off;
        off += (bytes + 255) & ~(size_t)255;
        return p;
    };
    bf16_t* xb      = (bf16_t*)alloc((size_t)T_ * D_ * 2);
    float*  logits  = (float*) alloc((size_t)T_ * E_ * 4);
    int*    tidx    = (int*)   alloc((size_t)T_ * TOPK_ * 4);
    float*  tw      = (float*) alloc((size_t)T_ * TOPK_ * 4);
    int*    cnt     = (int*)   alloc((size_t)E_ * 4);
    int*    slot_of = (int*)   alloc((size_t)T_ * TOPK_ * 4);
    bf16_t* xl      = (bf16_t*)alloc((size_t)T_ * DL_ * 2);
    bf16_t* buf     = (bf16_t*)alloc((size_t)E_ * CAP_ * DL_ * 2);
    bf16_t* H       = (bf16_t*)alloc((size_t)E_ * CAP_ * I_ * 2);
    float*  OUT     = (float*) alloc((size_t)E_ * CAP_ * DL_ * 4);
    bf16_t* routedb = (bf16_t*)alloc((size_t)T_ * DL_ * 2);
    bf16_t* S1      = (bf16_t*)alloc((size_t)T_ * SI_ * 2);
    (void)in_sizes; (void)n_in; (void)out_size; (void)ws_size;

    dim3 blk128(128), blk256(256), blk64(64);

    // 1) x -> bf16
    {
        int n4 = T_ * D_ / 4;
        cvt_f32_bf16<<<dim3((n4 + 255) / 256), blk256, 0, stream>>>(x, xb, n4);
    }
    // 2) gate logits: [T,E] = xb @ gate_w^T
    gemm_xwt<EPI_F32><<<dim3(E_ / 64, T_ / 128, 1), blk128, 0, stream>>>(
        xb, gate_w, logits, T_, E_, D_, 0, 0, 0, nullptr);
    // 3) routing (also zeroes cnt)
    routing_kernel<<<dim3(T_ / 256), blk256, 0, stream>>>(logits, gate_b, tidx, tw, cnt);
    // 4) fc1 latent projection: xl[T,DL] (bf16)
    gemm_xwt<EPI_BF16><<<dim3(DL_ / 64, T_ / 128, 1), blk128, 0, stream>>>(
        xb, fc1_w, xl, T_, DL_, D_, 0, 0, 0, nullptr);
    // 5) dispatch to per-expert buffers
    dispatch_kernel<<<dim3(T_ * TOPK_), blk64, 0, stream>>>(tidx, xl, buf, cnt, slot_of);
    // 6) expert GEMM 1: H = relu2(buf @ w1^T) per expert (count-gated tiles)
    gemm_xwt<EPI_RELU2_BF16><<<dim3(I_ / 64, CAP_ / 128, E_), blk128, 0, stream>>>(
        buf, w1, H, CAP_, I_, DL_,
        (long long)CAP_ * DL_, (long long)I_ * DL_, (long long)CAP_ * I_, cnt);
    // 7) expert GEMM 2: OUT = H @ w2^T per expert
    gemm_xwt<EPI_F32><<<dim3(DL_ / 64, CAP_ / 128, E_), blk128, 0, stream>>>(
        H, w2, OUT, CAP_, DL_, I_,
        (long long)CAP_ * I_, (long long)DL_ * I_, (long long)CAP_ * DL_, cnt);
    // 8) combine routed rows -> routedb (bf16)
    combine_kernel<<<dim3(T_), blk128, 0, stream>>>(OUT, slot_of, tw, routedb);
    // 9) shared expert: S1 = relu2(xb @ sw1^T)
    gemm_xwt<EPI_RELU2_BF16><<<dim3(SI_ / 64, T_ / 128, 1), blk128, 0, stream>>>(
        xb, sw1, S1, T_, SI_, D_, 0, 0, 0, nullptr);
    // 10) d_out = S1 @ sw2^T
    gemm_xwt<EPI_F32><<<dim3(D_ / 64, T_ / 128, 1), blk128, 0, stream>>>(
        S1, sw2, out, T_, D_, SI_, 0, 0, 0, nullptr);
    // 11) d_out += routedb @ fc2_w^T
    gemm_xwt<EPI_ADD_F32><<<dim3(D_ / 64, T_ / 128, 1), blk128, 0, stream>>>(
        routedb, fc2_w, out, T_, D_, DL_, 0, 0, 0, nullptr);
}
